// LSTMDecoder_24532853195105
// MI455X (gfx1250) — matrix operand order repaired and checked by reading, not compiled, not run on hardware
//
#include <hip/hip_runtime.h>
#include <hip/hip_bf16.h>

// Problem constants (from reference): B=256, H=512, O=7, L=2, T=64
#define B_ 256
#define H_ 512
#define L_ 2
#define O_ 7
#define T_ 64
#define G_ 2048   // 4*H gates
#define K_ 1024   // 2*H concat(x,h)

typedef __attribute__((ext_vector_type(16))) __bf16 v16bf;
typedef __attribute__((ext_vector_type(8)))  __bf16 v8bf;
typedef __attribute__((ext_vector_type(8)))  float  v8f;
typedef __attribute__((ext_vector_type(4)))  float  v4f;
typedef __attribute__((ext_vector_type(4)))  unsigned int v4u;
typedef __attribute__((ext_vector_type(4)))  int    v4i;
typedef __attribute__((ext_vector_type(8)))  int    v8i;

#if defined(__gfx1250__) && __has_builtin(__builtin_amdgcn_tensor_load_to_lds)
#define USE_TDM 1
#else
#define USE_TDM 0
#endif

// CDNA5 has a hardware V_TANH_F32 (TRANS32). Use it when the builtin exists.
#if defined(__gfx1250__) && __has_builtin(__builtin_amdgcn_tanhf)
__device__ __forceinline__ float fast_tanh(float x) { return __builtin_amdgcn_tanhf(x); }
#elif defined(__gfx1250__) && __has_builtin(__builtin_amdgcn_tanh_f32)
__device__ __forceinline__ float fast_tanh(float x) { return __builtin_amdgcn_tanh_f32(x); }
#else
__device__ __forceinline__ float fast_tanh(float x) { return tanhf(x); }
#endif
// sigmoid via tanh: one TRANS op + 2 cheap VALU ops
__device__ __forceinline__ float fast_sigm(float x) {
  return 0.5f * fast_tanh(0.5f * x) + 0.5f;
}

// ---------------------------------------------------------------------------
// Prep: gate-interleaved weight reorder + bf16 convert.
// Wt2 row r = j*4+g  (j=0..511 hidden unit, g=0..3 gate i/f/g/o) holds
//   [ W_ih[l][g*512+j][:], W_hh[l][g*512+j][:] ]  (K=1024 contiguous).
// A 256-wide N-tile then contains complete {i,f,g,o} quads for 64 j-units,
// allowing the LSTM cell to be fused into the GEMM epilogue.
// ---------------------------------------------------------------------------
__global__ void k_prep_weights(const float* __restrict__ W_ih,
                               const float* __restrict__ W_hh,
                               __bf16* __restrict__ Wt2) {
  int i = blockIdx.x * blockDim.x + threadIdx.x;   // over L_*G_*K_ = 2^22
  int l = i >> 21;                                 // G_*K_ = 2^21
  int r = (i >> 10) & (G_ - 1);                    // reordered row
  int k = i & (K_ - 1);
  int j = r >> 2;
  int g = r & 3;
  int srow = l * G_ + g * H_ + j;                  // original gate row
  float v = (k < H_) ? W_ih[srow * H_ + k]
                     : W_hh[srow * H_ + (k - H_)];
  Wt2[i] = (__bf16)v;
}

// ---------------------------------------------------------------------------
// Prep: state init. h0/c0 -> hT/cT output regions (running state lives there),
// h0 bf16 -> A[l] second half, x0 bf16 -> A0 first half,
// biasR[l][j*4+g] = b_ih[l][g*512+j] + b_hh[l][g*512+j]  (float4 per j).
// ---------------------------------------------------------------------------
__global__ void k_prep_state(const float* __restrict__ x,
                             const float* __restrict__ h0,
                             const float* __restrict__ c0,
                             const float* __restrict__ b_ih,
                             const float* __restrict__ b_hh,
                             __bf16* __restrict__ A0,
                             __bf16* __restrict__ A1,
                             float* __restrict__ biasR,
                             float* __restrict__ hT,
                             float* __restrict__ cT) {
  int i = blockIdx.x * blockDim.x + threadIdx.x;   // over L_*B_*H_ = 262144
  int l = i >> 17;                                 // B_*H_ = 2^17
  int r = i & ((1 << 17) - 1);
  int b = r >> 9;
  int k = r & (H_ - 1);
  float h = h0[i];
  hT[i] = h;
  cT[i] = c0[i];
  __bf16 hb = (__bf16)h;
  (l ? A1 : A0)[b * K_ + H_ + k] = hb;             // A[l][:, H:] = h0[l]
  if (l == 0) {
    A0[b * K_ + k] = (__bf16)x[b * H_ + k];        // A0[:, :H] = x[:,0,:]
  }
  if (i < L_ * G_) {                               // reordered bias
    int bl = i >> 11;
    int br = i & (G_ - 1);
    int bj = br >> 2;
    int bg = br & 3;
    biasR[i] = b_ih[bl * G_ + bg * H_ + bj] + b_hh[bl * G_ + bg * H_ + bj];
  }
}

// ---------------------------------------------------------------------------
// TDM: one tensor_load_to_lds moves a (tile_rows x 64)-bf16 tile from a
// row-major (tensor_rows x 1024) tensor into LDS, with HW padding of 4 DWORDs
// after every 32 DWORDs -> LDS row stride 144B = 72 halves (matches LDP).
// Descriptor packing per CDNA5 ISA.
// ---------------------------------------------------------------------------
#define MT 64
#define NT 256
#define KC 64
#define LDP 72
#define CS 260     // f32 LDS stride for the fused-cell C tile (64 x 256)

#if USE_TDM
__device__ __forceinline__ void tdm_load_tile(unsigned lds_addr,
                                              const __bf16* gptr,
                                              unsigned tensor_rows,
                                              unsigned tile_rows) {
  unsigned long long ga = (unsigned long long)(uintptr_t)gptr;
  // group0: count=1 | lds_addr | global_addr[56:0] | type=2
  v4u g0 = { 1u,
             lds_addr,
             (unsigned)ga,
             ((unsigned)((ga >> 32) & 0x01ffffffu)) | 0x80000000u };
  // group1: data_size=2B, pad_enable, pad_interval=32DW(code 4), pad_amount=4DW(code 3)
  //         tensor_dim0=1024, tensor_dim1=tensor_rows, tile_dim0=64(KC),
  //         tile_dim1=tile_rows, tile_dim2=0, tensor_dim0_stride=1024
  v8i g1 = { (int)((1u << 16) | (1u << 20) | (4u << 22) | (3u << 25)),
             (int)(1024u << 16),          // tensor_dim0[15:0] << 16
             (int)(tensor_rows << 16),    // tensor_dim0[31:16]=0 | tensor_dim1[15:0]
             (int)(64u << 16),            // tensor_dim1[31:16]=0 | tile_dim0
             (int)tile_rows,              // tile_dim1 | tile_dim2=0
             1024,                        // tensor_dim0_stride[31:0]
             0, 0 };
  v4i gz = { 0, 0, 0, 0 };
#if defined(__clang_major__) && (__clang_major__ >= 23)
  v8i gz8 = { 0, 0, 0, 0, 0, 0, 0, 0 };
  __builtin_amdgcn_tensor_load_to_lds(g0, g1, gz, gz, gz8, 0);
#else
  __builtin_amdgcn_tensor_load_to_lds(g0, g1, gz, gz, 0);
#endif
}
#endif

// ---------------------------------------------------------------------------
// Fused GEMM + LSTM cell.
// GEMM: C(256x2048,f32) = A(256x1024,bf16) * Wt2(2048x1024,bf16)^T with the
// gate-interleaved column order. Block: 256 threads = 8 waves (wave32),
// block tile M=64,N=256; wave tile 32x64 = 2x4 WMMA 16x16 accumulators.
// K staged through double-buffered LDS (TDM + TENSORcnt when available).
// Epilogue: accumulators -> LDS (f32), then per-(b,j) {i,f,g,o} quads are one
// ds_load_b128; cell math (hardware v_tanh) updates c/h in place and emits
// bf16 h for the next GEMM's A buffers. No gates round-trip to global memory.
// ---------------------------------------------------------------------------
__global__ __launch_bounds__(256) void k_gemm_cell(
    const __bf16* __restrict__ A,       // B_ x K_ (this layer's concat input)
    const __bf16* __restrict__ W,       // G_ x K_ (gate-interleaved rows)
    const float* __restrict__ biasR,    // H_ x 4 (this layer)
    float* __restrict__ c,              // B_ x H_ (layer slice of cT)
    float* __restrict__ h,              // B_ x H_ (layer slice of hT)
    __bf16* __restrict__ a_self,        // A_l + H_   (own second half)
    __bf16* __restrict__ a_other)       // A_other    (consumer first half)
{
  // 92160B: double-buffered bf16 tiles; aliased as 64x260 f32 C tile (66560B).
  __shared__ __align__(16) char smem[2 * MT * LDP * 2 + 2 * NT * LDP * 2];
  __bf16 (*sA)[MT * LDP] = (__bf16 (*)[MT * LDP])smem;
  __bf16 (*sB)[NT * LDP] = (__bf16 (*)[NT * LDP])(smem + 2 * MT * LDP * 2);
  float* sC = (float*)smem;

  const int tid  = threadIdx.x;
  const int lane = tid & 31;
  const int wave = tid >> 5;
  const int wm   = wave & 1;        // 2 waves across M
  const int wn   = wave >> 1;       // 4 waves across N
  const int m0   = (blockIdx.x & 3) * MT;
  const int n0   = (blockIdx.x >> 2) * NT;
  const int j0   = n0 >> 2;         // first hidden unit of this tile

  v8f acc[2][4];
  #pragma unroll
  for (int mt = 0; mt < 2; ++mt)
    #pragma unroll
    for (int nt = 0; nt < 4; ++nt)
      #pragma unroll
      for (int q = 0; q < 8; ++q) acc[mt][nt][q] = 0.0f;

#if USE_TDM
  if (wave == 0) {       // EXEC is ignored by TDM; issue once per workgroup
    tdm_load_tile((unsigned)(uintptr_t)&sA[0][0], A + m0 * K_, B_, MT);
    tdm_load_tile((unsigned)(uintptr_t)&sB[0][0], W + n0 * K_, G_, NT);
  }
#else
  const int arow = tid >> 2;
  const int achk = (tid & 3) * 8;
#endif

  const int NCHUNK = K_ / KC;        // 16
  for (int cc0 = 0; cc0 < NCHUNK; ++cc0) {
    const int cur = cc0 & 1;
    const int k0  = cc0 * KC;

#if USE_TDM
    if (wave == 0) {
      if (cc0 + 1 < NCHUNK) {
        const int nxt = (cc0 + 1) & 1;
        tdm_load_tile((unsigned)(uintptr_t)&sA[nxt][0], A + m0 * K_ + k0 + KC, B_, MT);
        tdm_load_tile((unsigned)(uintptr_t)&sB[nxt][0], W + n0 * K_ + k0 + KC, G_, NT);
        // TDM completes in-order per wave: <=2 outstanding => chunk cc0's pair landed
        __builtin_amdgcn_s_wait_tensorcnt(2);
      } else {
        __builtin_amdgcn_s_wait_tensorcnt(0);
      }
    }
    __syncthreads();     // publish LDS writes of chunk cc0 to all waves
#else
    *(v8bf*)&sA[cur][arow * LDP + achk] =
        *(const v8bf*)&A[(m0 + arow) * K_ + k0 + achk];
    #pragma unroll
    for (int q = 0; q < 4; ++q)
      *(v8bf*)&sB[cur][tid * LDP + q * 8] =
          *(const v8bf*)&W[(n0 + tid) * K_ + k0 + q * 8];
    __syncthreads();
#endif

    #pragma unroll
    for (int kk = 0; kk < KC; kk += 32) {
      v16bf af[2], bfm[4];
      // A frag 16x32: lanes 0-15 hold K[0..7]++K[16..23], lanes 16-31 K[8..15]++K[24..31]
      const int alo = kk + ((lane & 16) ? 8 : 0);
      #pragma unroll
      for (int mt = 0; mt < 2; ++mt) {
        const __bf16* p = &sA[cur][(wm * 32 + mt * 16 + (lane & 15)) * LDP + alo];
        union { v16bf v; v8bf hh[2]; } u;
        u.hh[0] = *(const v8bf*)p;
        u.hh[1] = *(const v8bf*)(p + 16);
        af[mt] = u.v;
      }
      // B frag 32x16: lane holds one N-column (= weight row), same K split as A
      const int blo = kk + ((lane & 16) ? 8 : 0);
      #pragma unroll
      for (int nt = 0; nt < 4; ++nt) {
        const __bf16* p = &sB[cur][(wn * 64 + nt * 16 + (lane & 15)) * LDP + blo];
        union { v16bf v; v8bf hh[2]; } u;
        u.hh[0] = *(const v8bf*)p;
        u.hh[1] = *(const v8bf*)(p + 16);
        bfm[nt] = u.v;
      }
      #pragma unroll
      for (int mt = 0; mt < 2; ++mt)
        #pragma unroll
        for (int nt = 0; nt < 4; ++nt)
          acc[mt][nt] = __builtin_amdgcn_wmma_f32_16x16x32_bf16(
              false, af[mt], false, bfm[nt], (short)0, acc[mt][nt],
              false, false);
    }
    __syncthreads();     // all waves done reading before buffer is re-filled
  }

  // ---- Epilogue 1: accumulators -> LDS (C layout: VGPR v -> M=v / v+8) ----
  #pragma unroll
  for (int mt = 0; mt < 2; ++mt)
    #pragma unroll
    for (int nt = 0; nt < 4; ++nt) {
      const int rowL = wm * 32 + mt * 16 + ((lane & 16) ? 8 : 0);
      const int colL = wn * 64 + nt * 16 + (lane & 15);
      #pragma unroll
      for (int v = 0; v < 8; ++v)
        sC[(rowL + v) * CS + colL] = acc[mt][nt][v];
    }
  __syncthreads();

  // ---- Epilogue 2: fused LSTM cell over the 64x64 (b,j) tile ----
  #pragma unroll 4
  for (int p = 0; p < (MT * (NT / 4)) / 256; ++p) {   // 16 iters
    const int idx = tid + p * 256;
    const int ml  = idx >> 6;          // local row (batch)
    const int jl  = idx & 63;          // local hidden unit
    v4f gv = *(const v4f*)&sC[ml * CS + jl * 4];      // {i,f,g,o} pre-activation
    v4f bv = *(const v4f*)&biasR[(j0 + jl) * 4];
    float gi = fast_sigm(gv[0] + bv[0]);
    float gf = fast_sigm(gv[1] + bv[1]);
    float gg = fast_tanh(gv[2] + bv[2]);
    float go = fast_sigm(gv[3] + bv[3]);
    const int b  = m0 + ml;
    const int j  = j0 + jl;
    const int si = b * H_ + j;
    float cn = gf * c[si] + gi * gg;
    c[si] = cn;
    float hn = go * fast_tanh(cn);
    h[si] = hn;
    __bf16 hb = (__bf16)hn;
    a_self [b * K_ + j] = hb;
    a_other[b * K_ + j] = hb;
  }
}

// ---------------------------------------------------------------------------
// FC head: y[b,o] = h1[b,:]·W_fc[o,:] + b_fc[o]  (256x7x512, tiny)
// One thread per (b,o) pair.
// ---------------------------------------------------------------------------
__global__ void k_fc(const float* __restrict__ h,
                     const float* __restrict__ Wfc,
                     const float* __restrict__ bfc,
                     float* __restrict__ y) {     // points at decoded + t*O_
  int b = threadIdx.x;
  int o = blockIdx.x;
  const float* hb = &h[b * H_];
  const float* wo = &Wfc[o * H_];
  float accv = bfc[o];
  for (int k = 0; k < H_; k += 4) {
    accv += hb[k]     * wo[k]     + hb[k + 1] * wo[k + 1]
          + hb[k + 2] * wo[k + 2] + hb[k + 3] * wo[k + 3];
  }
  y[b * (T_ * O_) + o] = accv;
}

// ---------------------------------------------------------------------------
extern "C" void kernel_launch(void* const* d_in, const int* in_sizes, int n_in,
                              void* d_out, int out_size, void* d_ws, size_t ws_size,
                              hipStream_t stream) {
  const float* x    = (const float*)d_in[0];
  const float* h0   = (const float*)d_in[1];
  const float* c0   = (const float*)d_in[2];
  const float* W_ih = (const float*)d_in[3];
  const float* W_hh = (const float*)d_in[4];
  const float* b_ih = (const float*)d_in[5];
  const float* b_hh = (const float*)d_in[6];
  const float* W_fc = (const float*)d_in[7];
  const float* b_fc = (const float*)d_in[8];
  // d_in[9] = seq_len; fixed to T_=64 per problem constants.

  char* ws = (char*)d_ws;
  __bf16* Wt2   = (__bf16*)ws;                                  // 2*2048*1024*2 = 8 MB
  __bf16* A0    = (__bf16*)(ws + 8388608);                      // 256*1024*2 = 512 KB
  __bf16* A1    = (__bf16*)(ws + 8388608 + 524288);             // 512 KB
  float*  biasR = (float*)(ws + 9437184);                       // 2*2048*4

  float* out     = (float*)d_out;
  float* decoded = out;                         // B*T*O
  float* hT      = out + B_ * T_ * O_;          // L*B*H (running h lives here)
  float* cT      = hT + L_ * B_ * H_;           // L*B*H (running c lives here)

  k_prep_weights<<<dim3((L_ * G_ * K_) / 256), 256, 0, stream>>>(W_ih, W_hh, Wt2);
  k_prep_state  <<<dim3((L_ * B_ * H_) / 256), 256, 0, stream>>>(
      x, h0, c0, b_ih, b_hh, A0, A1, biasR, hT, cT);

  for (int t = 0; t < T_; ++t) {
    for (int l = 0; l < L_; ++l) {
      __bf16* Al = l ? A1 : A0;
      __bf16* Ao = l ? A0 : A1;
      k_gemm_cell<<<dim3(32), 256, 0, stream>>>(
          Al, Wt2 + (size_t)l * G_ * K_, biasR + l * G_,
          cT + l * B_ * H_, hT + l * B_ * H_,
          Al + H_, Ao);
    }
    k_fc<<<dim3(O_), 256, 0, stream>>>(hT + B_ * H_, W_fc, b_fc, decoded + t * O_);
  }
}
